// Generator_17437567222068
// MI455X (gfx1250) — matrix operand order repaired and checked by reading, not compiled, not run on hardware
//
#include <hip/hip_runtime.h>
#include <math.h>

// ---------------------------------------------------------------------------
// MAC-network forward for MI455X (gfx1250).
// GEMMs: f16 pre-packed fragments -> branch-free 4x-register-blocked
// v_wmma_f32_16x16x32_f16 loop (16x64 C strip per wave).
// ---------------------------------------------------------------------------

#define HH     300
#define DD     600
#define G3H    900
#define VV     19495
#define LSEQ   128
#define TDEC   1024
#define NTURNS 4
#define NMAC   7

typedef __attribute__((ext_vector_type(16))) _Float16 v16h;
typedef __attribute__((ext_vector_type(8)))  _Float16 v8h;
typedef __attribute__((ext_vector_type(8)))  float    v8f;

static inline int cdiv(int a, int b) { return (a + b - 1) / b; }
static inline int pad32(int k) { return (k + 31) & ~31; }
static inline int pad64(int n) { return (n + 63) & ~63; }

// ======================= f32 -> f16 packers ================================
// A[M,K] (lda) -> Ap[M,Kp] zero-padded f16, row-major.
__global__ void g17_packa(const float* __restrict__ A, _Float16* __restrict__ Ap,
                          int M, int K, int lda, int Kp)
{
    const int idx = blockIdx.x * blockDim.x + threadIdx.x;
    if (idx >= M * Kp) return;
    const int m = idx / Kp, k = idx - m * Kp;
    Ap[idx] = (k < K) ? (_Float16)A[(size_t)m * lda + k] : (_Float16)0.0f;
}
// B[K,N] (ldb) -> Bt[Np,Kp] zero-padded f16: Bt[n*Kp+k] = B[k*ldb+n].
__global__ void g17_packbt(const float* __restrict__ B, _Float16* __restrict__ Bt,
                           int K, int N, int ldb, int Kp, int Np)
{
    const int idx = blockIdx.x * blockDim.x + threadIdx.x;
    if (idx >= Np * Kp) return;
    const int n = idx / Kp, k = idx - n * Kp;
    Bt[idx] = (k < K && n < N) ? (_Float16)B[(size_t)k * ldb + n] : (_Float16)0.0f;
}

// ======================= WMMA GEMM (16x64 strip per wave) ==================
// C[M,N] = act(A @ B + bias); Ap f16 [M,Kp] row-major, Bt f16 [Np64,Kp]
// (transposed). Kp mult of 32, M mult of 16, Np64 = pad64(N).
// Block = 256 threads = 8 waves; each wave: 1 M-tile x 4 N-tiles.
// grid = (Np64/64, ceil(M/128)).
// A fragment per lane (row mr): halfs [k0+16g .. +15]  (contiguous in Ap).
// B fragment per lane (col nc): halfs [k0+16g .. +15]  (contiguous in Bt).
__global__ void g17_gemm(const _Float16* __restrict__ Ap, const _Float16* __restrict__ Bt,
                         const float* __restrict__ bias, float* __restrict__ C,
                         int M, int N, int Kp, int ldc, int act)
{
    const int wave = threadIdx.x >> 5;
    const int lane = threadIdx.x & 31;
    const int tm = blockIdx.y * 8 + wave;
    const int m0 = tm * 16;
    if (m0 >= M) return;                       // wave-uniform
    const int n0 = blockIdx.x * 64;
    const int g  = lane >> 4;
    const int mr = m0 + (lane & 15);
    const _Float16* ap = Ap + (size_t)mr * Kp + 16 * g;
    const _Float16* bp = Bt + (size_t)(n0 + (lane & 15)) * Kp + 16 * g;
    const size_t bstep = (size_t)16 * Kp;      // next 16-col subtile
    v8f acc0 = {}, acc1 = {}, acc2 = {}, acc3 = {};
    for (int k0 = 0; k0 < Kp; k0 += 32) {
        const v8h alo = *(const v8h*)(ap + k0);
        const v8h ahi = *(const v8h*)(ap + k0 + 8);
        const v16h a = __builtin_shufflevector(alo, ahi,
            0, 1, 2, 3, 4, 5, 6, 7, 8, 9, 10, 11, 12, 13, 14, 15);
        const v16h b0 = *(const v16h*)(bp + k0);
        const v16h b1 = *(const v16h*)(bp + bstep + k0);
        const v16h b2 = *(const v16h*)(bp + 2 * bstep + k0);
        const v16h b3 = *(const v16h*)(bp + 3 * bstep + k0);
        acc0 = __builtin_amdgcn_wmma_f32_16x16x32_f16(false, a, false, b0, (short)0, acc0, false, false);
        acc1 = __builtin_amdgcn_wmma_f32_16x16x32_f16(false, a, false, b1, (short)0, acc1, false, false);
        acc2 = __builtin_amdgcn_wmma_f32_16x16x32_f16(false, a, false, b2, (short)0, acc2, false, false);
        acc3 = __builtin_amdgcn_wmma_f32_16x16x32_f16(false, a, false, b3, (short)0, acc3, false, false);
    }
    v8f accs[4] = { acc0, acc1, acc2, acc3 };
#pragma unroll
    for (int s = 0; s < 4; ++s) {
        const int nc = n0 + 16 * s + (lane & 15);
        if (nc < N) {
            const float bv = (bias != nullptr) ? bias[nc] : 0.0f;
#pragma unroll
            for (int r = 0; r < 8; ++r) {
                const int mm = m0 + r + 8 * g;
                float vv = accs[s][r] + bv;
                if (act == 1) vv = tanhf(vv);
                C[(size_t)mm * ldc + nc] = vv;
            }
        }
    }
}

// ======================= Embedding gather ==================================
__global__ void g17_gather(const float* __restrict__ emb, const int* __restrict__ idx,
                           float* __restrict__ out)
{
    const int r = blockIdx.x;
    const int row = idx[r];
    for (int h = threadIdx.x; h < HH; h += blockDim.x)
        out[(size_t)r * HH + h] = emb[(size_t)row * HH + h];
}

// ======================= GRU sequential scan ===============================
__global__ void g17_gru_scan(const float* __restrict__ gi, const float* __restrict__ Whh,
                             const float* __restrict__ bhh, const float* __restrict__ h0,
                             float* __restrict__ out, int L, int out_stride, int out_off)
{
    __shared__ float h[HH];
    __shared__ float gh[G3H];
    const int t = threadIdx.x;
    if (t < HH) h[t] = h0 ? h0[t] : 0.0f;
    __syncthreads();
    for (int l = 0; l < L; ++l) {
        if (t < G3H) {
            float s = bhh[t];
            for (int k = 0; k < HH; ++k) s += h[k] * Whh[(size_t)k * G3H + t];
            gh[t] = s;
        }
        __syncthreads();
        if (t < HH) {
            const float ir  = gi[(size_t)l * G3H + t];
            const float iz  = gi[(size_t)l * G3H + HH + t];
            const float inn = gi[(size_t)l * G3H + 2 * HH + t];
            const float r = 1.0f / (1.0f + expf(-(ir + gh[t])));
            const float z = 1.0f / (1.0f + expf(-(iz + gh[HH + t])));
            const float n = tanhf(inn + r * gh[2 * HH + t]);
            const float hn = (1.0f - z) * n + z * h[t];
            h[t] = hn;
            out[(size_t)l * out_stride + out_off + t] = hn;
        }
        __syncthreads();
    }
}

// ======================= matvec y = act((x1++x2) @ W + b) ==================
// act: 0 none, 1 tanh, 3 sigmoid(tanh(.))
__global__ void g17_matvec(const float* __restrict__ x1, int K1,
                           const float* __restrict__ x2, int K2,
                           const float* __restrict__ W, const float* __restrict__ bias,
                           float bias_const, float* __restrict__ y, int N, int act)
{
    extern __shared__ float xs[];
    const int K = K1 + K2;
    for (int k = threadIdx.x; k < K; k += blockDim.x)
        xs[k] = (k < K1) ? x1[k] : x2[k - K1];
    __syncthreads();
    const int n = blockIdx.x * blockDim.x + threadIdx.x;
    if (n >= N) return;
    float s = bias ? bias[n] : bias_const;
    for (int k = 0; k < K; ++k) s += xs[k] * W[(size_t)k * N + n];
    if (act == 1) s = tanhf(s);
    else if (act == 3) { s = tanhf(s); s = 1.0f / (1.0f + expf(-s)); }
    y[n] = s;
}

// ======================= attention scores ==================================
__global__ void g17_scores(const float* __restrict__ Rows, int ld,
                           const float* __restrict__ x, const float* __restrict__ w,
                           const float* __restrict__ b, float* __restrict__ out,
                           int L, int Dd)
{
    const int l = blockIdx.x * blockDim.x + threadIdx.x;
    if (l >= L) return;
    float s = b[0];
    for (int d = 0; d < Dd; ++d) s += x[d] * Rows[(size_t)l * ld + d] * w[d];
    out[l] = s;
}

// ======================= softmax (in place, single block of 128) ===========
__global__ void g17_softmax(float* __restrict__ v, int n)
{
    __shared__ float red[128];
    const int t = threadIdx.x;
    float m = -1e30f;
    for (int i = t; i < n; i += blockDim.x) m = fmaxf(m, v[i]);
    red[t] = m; __syncthreads();
    for (int s = blockDim.x / 2; s > 0; s >>= 1) {
        if (t < s) red[t] = fmaxf(red[t], red[t + s]);
        __syncthreads();
    }
    m = red[0]; __syncthreads();
    float sum = 0.0f;
    for (int i = t; i < n; i += blockDim.x) { const float e = expf(v[i] - m); v[i] = e; sum += e; }
    red[t] = sum; __syncthreads();
    for (int s = blockDim.x / 2; s > 0; s >>= 1) {
        if (t < s) red[t] += red[t + s];
        __syncthreads();
    }
    const float inv = 1.0f / red[0];
    for (int i = t; i < n; i += blockDim.x) v[i] *= inv;
}

// ======================= out[d] = tanh( (1/L) sum_l a[l]*Rows[l,d] ) =======
__global__ void g17_wmean(const float* __restrict__ Rows, int ld,
                          const float* __restrict__ a, float* __restrict__ out,
                          int L, int Dd)
{
    const int d = blockIdx.x * blockDim.x + threadIdx.x;
    if (d >= Dd) return;
    float s = 0.0f;
    for (int l = 0; l < L; ++l) s += a[l] * Rows[(size_t)l * ld + d];
    out[d] = tanhf(s / (float)L);
}

// ======================= read-unit concat builder ==========================
__global__ void g17_buildcat(const float* __restrict__ kbp, const float* __restrict__ kb,
                             const float* __restrict__ m_, float* __restrict__ A)
{
    const int l = blockIdx.x;
    for (int d = threadIdx.x; d < DD; d += blockDim.x) {
        A[(size_t)l * 2 * DD + d]      = kbp[(size_t)l * DD + d] * m_[d];
        A[(size_t)l * 2 * DD + DD + d] = kb[(size_t)l * DD + d];
    }
}

// ======================= gated combine =====================================
__global__ void g17_gate(const float* __restrict__ g, const float* __restrict__ mp,
                         const float* __restrict__ mn, float* __restrict__ mo, int n)
{
    const int i = blockIdx.x * blockDim.x + threadIdx.x;
    if (i < n) mo[i] = g[i] * mp[i] + (1.0f - g[i]) * mn[i];
}

// ======================= copies ============================================
__global__ void g17_copy(const float* __restrict__ s, float* __restrict__ d, int n)
{
    const int i = blockIdx.x * blockDim.x + threadIdx.x;
    if (i < n) d[i] = s[i];
}
__global__ void g17_tanhcopy(const float* __restrict__ s, float* __restrict__ d, int n)
{
    const int i = blockIdx.x * blockDim.x + threadIdx.x;
    if (i < n) d[i] = tanhf(s[i]);
}

// ===========================================================================
extern "C" void kernel_launch(void* const* d_in, const int* in_sizes, int n_in,
                              void* d_out, int out_size, void* d_ws, size_t ws_size,
                              hipStream_t stream)
{
    (void)in_sizes; (void)n_in; (void)out_size; (void)ws_size;

    const float* emb      = (const float*)d_in[0];
    const float* kb_Wih0  = (const float*)d_in[1];
    const float* kb_Whh0  = (const float*)d_in[2];
    const float* kb_bih0  = (const float*)d_in[3];
    const float* kb_bhh0  = (const float*)d_in[4];
    const float* kb_Wih1  = (const float*)d_in[5];
    const float* kb_Whh1  = (const float*)d_in[6];
    const float* kb_bih1  = (const float*)d_in[7];
    const float* kb_bhh1  = (const float*)d_in[8];
    const float* dec_Wih  = (const float*)d_in[9];
    const float* dec_Whh  = (const float*)d_in[10];
    const float* dec_bih  = (const float*)d_in[11];
    const float* dec_bhh  = (const float*)d_in[12];
    const float* q_W      = (const float*)d_in[13];
    const float* q_b      = (const float*)d_in[14];
    const float* state_W  = (const float*)d_in[15];
    const float* state_b  = (const float*)d_in[16];
    const float* out_W    = (const float*)d_in[17];
    const float* out_b    = (const float*)d_in[18];
    const float* c_W_2dd  = (const float*)d_in[19];
    const float* c_b_d    = (const float*)d_in[20];
    const float* c_W_d1   = (const float*)d_in[21];
    const float* c_b_1    = (const float*)d_in[22];
    const float* r_W1     = (const float*)d_in[23];
    const float* r_b1     = (const float*)d_in[24];
    const float* r_W2     = (const float*)d_in[25];
    const float* r_b2     = (const float*)d_in[26];
    const float* r_W3     = (const float*)d_in[27];
    const float* r_b3     = (const float*)d_in[28];
    const float* r_W4     = (const float*)d_in[29];
    const float* r_b4     = (const float*)d_in[30];
    const float* w_W1     = (const float*)d_in[31];
    const float* w_b1     = (const float*)d_in[32];
    const float* w_W2     = (const float*)d_in[33];
    const float* w_b2     = (const float*)d_in[34];
    const float* w_W3     = (const float*)d_in[35];
    const float* w_b3     = (const float*)d_in[36];
    const float* w_W4     = (const float*)d_in[37];
    const float* w_b4     = (const float*)d_in[38];
    const float* Wdd_all  = (const float*)d_in[39];
    const float* init_m   = (const float*)d_in[40];
    const float* init_c   = (const float*)d_in[41];
    const int*   con_kb   = (const int*)d_in[42];
    const int*   con_q    = (const int*)d_in[43];
    const int*   tgt      = (const int*)d_in[44];
    float* out = (float*)d_out;

    // ---- workspace carve-up: f32 region then 64B-aligned f16 region ----
    float* w = (float*)d_ws;
    size_t off = 0;
    auto alloc = [&](size_t n) { float* p = w + off; off += n; return p; };
    float* f_kbx  = alloc((size_t)LSEQ * HH);
    float* f_qx   = alloc((size_t)LSEQ * HH);
    float* f_gi   = alloc((size_t)TDEC * G3H);
    float* f_kb   = alloc((size_t)LSEQ * DD);
    float* f_cw   = alloc((size_t)LSEQ * DD);
    float* f_kbp  = alloc((size_t)LSEQ * DD);
    float* f_cat  = alloc((size_t)LSEQ * 2 * DD);
    float* f_imkb = alloc((size_t)LSEQ * DD);
    float* f_q    = alloc(DD);
    float* f_qi   = alloc(DD);
    float* f_cq   = alloc(DD);
    float* f_sc   = alloc(LSEQ);
    float* f_mr   = alloc(DD);
    float* f_mnew = alloc(DD);
    float* f_mw1  = alloc(DD);
    float* f_msa  = alloc(DD);
    float* f_mpp  = alloc(DD);
    float* f_g    = alloc(DD);
    float* f_wsc  = alloc(32);
    float* f_M    = alloc((size_t)(NTURNS * NMAC + 1) * DD);
    float* f_C    = alloc((size_t)(NTURNS * NMAC + 1) * DD);
    float* f_mid  = alloc(4 * DD);
    float* f_h0   = alloc(HH);
    float* f_tgtx = alloc((size_t)TDEC * HH);
    float* f_hs   = alloc((size_t)TDEC * HH);
    off = (off + 15) & ~(size_t)15;                 // 64B-align f16 region
    _Float16* hbase = (_Float16*)(w + off);
    size_t hoff = 0;
    auto halloc = [&](size_t n) { _Float16* p = hbase + hoff; hoff += n; return p; };

    const int KP_H  = pad32(HH);        // 320
    const int KP_D  = pad32(DD);        // 608
    const int KP_2D = pad32(2 * DD);    // 1216
    const int NP_3H = pad64(G3H);       // 960
    const int NP_D  = pad64(DD);        // 640
    const int NP_V  = pad64(VV);        // 19520

    _Float16* h_wih0 = halloc((size_t)NP_3H * KP_H);
    _Float16* h_wih1 = halloc((size_t)NP_3H * KP_H);
    _Float16* h_rw2  = halloc((size_t)NP_D * KP_D);
    _Float16* h_rw3  = halloc((size_t)NP_D * KP_2D);
    _Float16* h_dwih = halloc((size_t)NP_3H * KP_H);
    _Float16* h_outw = halloc((size_t)NP_V * KP_H);
    _Float16* h_ap   = halloc((size_t)TDEC * KP_H); // reused A-pack (max size)

    const dim3 B256(256);
    const size_t sh600  = DD * sizeof(float);
    const size_t sh1200 = 2 * DD * sizeof(float);

    // ---- pack all weight (B-side) matrices once ----
    g17_packbt<<<cdiv(NP_3H * KP_H, 256), B256, 0, stream>>>(kb_Wih0, h_wih0, HH, G3H, G3H, KP_H, NP_3H);
    g17_packbt<<<cdiv(NP_3H * KP_H, 256), B256, 0, stream>>>(kb_Wih1, h_wih1, HH, G3H, G3H, KP_H, NP_3H);
    g17_packbt<<<cdiv(NP_D * KP_D, 256), B256, 0, stream>>>(r_W2, h_rw2, DD, DD, DD, KP_D, NP_D);
    g17_packbt<<<cdiv(NP_D * KP_2D, 256), B256, 0, stream>>>(r_W3, h_rw3, 2 * DD, DD, DD, KP_2D, NP_D);
    g17_packbt<<<cdiv(NP_3H * KP_H, 256), B256, 0, stream>>>(dec_Wih, h_dwih, HH, G3H, G3H, KP_H, NP_3H);
    g17_packbt<<<cdiv(NP_V * KP_H, 256), B256, 0, stream>>>(out_W, h_outw, HH, VV, VV, KP_H, NP_V);

    // history[0] = init
    g17_copy<<<cdiv(DD, 256), B256, 0, stream>>>(init_m, f_M, DD);
    g17_copy<<<cdiv(DD, 256), B256, 0, stream>>>(init_c, f_C, DD);

    for (int i = 0; i < NTURNS; ++i) {
        // ---- encode kb ----
        g17_gather<<<LSEQ, 320, 0, stream>>>(emb, con_kb + i * LSEQ, f_kbx);
        g17_packa<<<cdiv(LSEQ * KP_H, 256), B256, 0, stream>>>(f_kbx, h_ap, LSEQ, HH, HH, KP_H);
        g17_gemm<<<dim3(NP_3H / 64, 1), B256, 0, stream>>>(h_ap, h_wih0, kb_bih0, f_gi, LSEQ, G3H, KP_H, G3H, 0);
        g17_gru_scan<<<1, 960, 0, stream>>>(f_gi, kb_Whh0, kb_bhh0, nullptr, f_kb, LSEQ, DD, 0);
        g17_gemm<<<dim3(NP_3H / 64, 1), B256, 0, stream>>>(h_ap, h_wih1, kb_bih1, f_gi, LSEQ, G3H, KP_H, G3H, 0);
        g17_gru_scan<<<1, 960, 0, stream>>>(f_gi, kb_Whh1, kb_bhh1, nullptr, f_kb, LSEQ, DD, HH);
        // ---- encode q (same encoder, per reference) ----
        g17_gather<<<LSEQ, 320, 0, stream>>>(emb, con_q + i * LSEQ, f_qx);
        g17_packa<<<cdiv(LSEQ * KP_H, 256), B256, 0, stream>>>(f_qx, h_ap, LSEQ, HH, HH, KP_H);
        g17_gemm<<<dim3(NP_3H / 64, 1), B256, 0, stream>>>(h_ap, h_wih0, kb_bih0, f_gi, LSEQ, G3H, KP_H, G3H, 0);
        g17_gru_scan<<<1, 960, 0, stream>>>(f_gi, kb_Whh0, kb_bhh0, nullptr, f_cw, LSEQ, DD, 0);
        g17_gemm<<<dim3(NP_3H / 64, 1), B256, 0, stream>>>(h_ap, h_wih1, kb_bih1, f_gi, LSEQ, G3H, KP_H, G3H, 0);
        g17_gru_scan<<<1, 960, 0, stream>>>(f_gi, kb_Whh1, kb_bhh1, nullptr, f_cw, LSEQ, DD, HH);
        // ---- q = [cw[0]@qW+qb , cw[-1]@qW+qb] ----
        g17_matvec<<<cdiv(HH, 256), B256, sh600, stream>>>(
            f_cw, DD, nullptr, 0, q_W, q_b, 0.0f, f_q, HH, 0);
        g17_matvec<<<cdiv(HH, 256), B256, sh600, stream>>>(
            f_cw + (size_t)(LSEQ - 1) * DD, DD, nullptr, 0, q_W, q_b, 0.0f, f_q + HH, HH, 0);
        // ---- kb_ = tanh(kb @ r_W2 + r_b2)  (turn-invariant, hoisted) ----
        g17_packa<<<cdiv(LSEQ * KP_D, 256), B256, 0, stream>>>(f_kb, h_ap, LSEQ, DD, DD, KP_D);
        g17_gemm<<<dim3(NP_D / 64, 1), B256, 0, stream>>>(h_ap, h_rw2, r_b2, f_kbp, LSEQ, DD, KP_D, DD, 1);

        for (int j = 0; j < NMAC; ++j) {
            const int t = i * NMAC + j;
            const float* Wdd = Wdd_all + (size_t)t * DD * DD;
            float* c_prev = f_C + (size_t)t * DD;
            float* m_prev = f_M + (size_t)t * DD;
            float* c_cur  = f_C + (size_t)(t + 1) * DD;
            float* m_cur  = f_M + (size_t)(t + 1) * DD;
            // ---- control ----
            g17_matvec<<<cdiv(DD, 256), B256, sh600, stream>>>(
                f_q, DD, nullptr, 0, Wdd, nullptr, 0.1f, f_qi, DD, 1);
            g17_matvec<<<cdiv(DD, 256), B256, sh1200, stream>>>(
                f_qi, DD, c_prev, DD, c_W_2dd, c_b_d, 0.0f, f_cq, DD, 1);
            g17_scores<<<1, 128, 0, stream>>>(f_cw, DD, f_cq, c_W_d1, c_b_1, f_sc, LSEQ, DD);
            g17_softmax<<<1, 128, 0, stream>>>(f_sc, LSEQ);
            g17_wmean<<<cdiv(DD, 256), B256, 0, stream>>>(f_cw, DD, f_sc, c_cur, LSEQ, DD);
            // ---- read ----
            g17_matvec<<<cdiv(DD, 256), B256, sh600, stream>>>(
                m_prev, DD, nullptr, 0, r_W1, r_b1, 0.0f, f_mr, DD, 1);
            g17_buildcat<<<LSEQ, B256, 0, stream>>>(f_kbp, f_kb, f_mr, f_cat);
            g17_packa<<<cdiv(LSEQ * KP_2D, 256), B256, 0, stream>>>(f_cat, h_ap, LSEQ, 2 * DD, 2 * DD, KP_2D);
            g17_gemm<<<dim3(NP_D / 64, 1), B256, 0, stream>>>(h_ap, h_rw3, r_b3, f_imkb, LSEQ, DD, KP_2D, DD, 1);
            g17_scores<<<1, 128, 0, stream>>>(f_imkb, DD, c_cur, r_W4, r_b4, f_sc, LSEQ, DD);
            g17_softmax<<<1, 128, 0, stream>>>(f_sc, LSEQ);
            g17_wmean<<<cdiv(DD, 256), B256, 0, stream>>>(f_kb, DD, f_sc, f_mnew, LSEQ, DD);
            // ---- write ----
            g17_matvec<<<cdiv(DD, 256), B256, sh1200, stream>>>(
                f_mnew, DD, m_prev, DD, w_W1, w_b1, 0.0f, f_mw1, DD, 1);
            g17_scores<<<1, 128, 0, stream>>>(f_C, DD, c_cur, w_W2, w_b2, f_wsc, t + 1, DD);
            g17_softmax<<<1, 128, 0, stream>>>(f_wsc, t + 1);
            g17_wmean<<<cdiv(DD, 256), B256, 0, stream>>>(f_M, DD, f_wsc, f_msa, t + 1, DD);
            g17_matvec<<<cdiv(DD, 256), B256, sh1200, stream>>>(
                f_msa, DD, f_mw1, DD, w_W3, w_b3, 0.0f, f_mpp, DD, 1);
            g17_matvec<<<cdiv(DD, 256), B256, sh600, stream>>>(
                c_cur, DD, nullptr, 0, w_W4, w_b4, 0.0f, f_g, DD, 3);
            g17_gate<<<cdiv(DD, 256), B256, 0, stream>>>(f_g, m_prev, f_mpp, m_cur, DD);
        }
    }

    // ---- mid = [kb[0], cw[0], tanh(q), tanh(m_last)] ----
    g17_copy<<<cdiv(DD, 256), B256, 0, stream>>>(f_kb, f_mid, DD);
    g17_copy<<<cdiv(DD, 256), B256, 0, stream>>>(f_cw, f_mid + DD, DD);
    g17_tanhcopy<<<cdiv(DD, 256), B256, 0, stream>>>(f_q, f_mid + 2 * DD, DD);
    g17_tanhcopy<<<cdiv(DD, 256), B256, 0, stream>>>(
        f_M + (size_t)(NTURNS * NMAC) * DD, f_mid + 3 * DD, DD);
    g17_matvec<<<cdiv(HH, 256), B256, 4 * DD * sizeof(float), stream>>>(
        f_mid, 4 * DD, nullptr, 0, state_W, state_b, 0.0f, f_h0, HH, 0);

    // ---- decoder ----
    g17_gather<<<TDEC, 320, 0, stream>>>(emb, tgt, f_tgtx);
    g17_packa<<<cdiv(TDEC * KP_H, 256), B256, 0, stream>>>(f_tgtx, h_ap, TDEC, HH, HH, KP_H);
    g17_gemm<<<dim3(NP_3H / 64, cdiv(TDEC, 128)), B256, 0, stream>>>(
        h_ap, h_dwih, dec_bih, f_gi, TDEC, G3H, KP_H, G3H, 0);
    g17_gru_scan<<<1, 960, 0, stream>>>(f_gi, dec_Whh, dec_bhh, f_h0, f_hs, TDEC, HH, 0);

    // ---- output projection [1024,300]@[300,19495] ----
    g17_packa<<<cdiv(TDEC * KP_H, 256), B256, 0, stream>>>(f_hs, h_ap, TDEC, HH, HH, KP_H);
    g17_gemm<<<dim3(NP_V / 64, cdiv(TDEC, 128)), B256, 0, stream>>>(
        h_ap, h_outw, out_b, out, TDEC, VV, KP_H, VV, 0);
}
